// GraphFusionNetwork_3908420240323
// MI455X (gfx1250) — matrix operand order repaired and checked by reading, not compiled, not run on hardware
//
#include <hip/hip_runtime.h>
#include <hip/hip_bf16.h>

typedef __bf16 bf16_t;
typedef __bf16 v8bf   __attribute__((ext_vector_type(8)));
typedef __bf16 v16bf  __attribute__((ext_vector_type(16)));
typedef float  v8f    __attribute__((ext_vector_type(8)));

#define Bsz 32
#define Dd  128
#define NTOT 4353   // ct(1024)|pet(1024)|clin(256)|gen(2048)|pat(1)

// ---------- helpers ----------
__device__ __forceinline__ unsigned f2ord(float f) {
    unsigned u = __float_as_uint(f);
    return (u & 0x80000000u) ? ~u : (u | 0x80000000u);
}
__device__ __forceinline__ float ord2f(unsigned s) {
    unsigned u = (s & 0x80000000u) ? (s & 0x7FFFFFFFu) : ~s;
    return __uint_as_float(u);
}
__device__ __forceinline__ float leaky(float x) { return x > 0.f ? x : 0.2f * x; }

// ---------- init / pack ----------
__global__ void zero_f32_kernel(float* p, size_t n) {
    size_t i = (size_t)blockIdx.x * blockDim.x + threadIdx.x;
    if (i < n) p[i] = 0.f;
}
__global__ void zero_u32_kernel(unsigned* p, size_t n) {
    size_t i = (size_t)blockIdx.x * blockDim.x + threadIdx.x;
    if (i < n) p[i] = 0u;   // ordered-encoding minimum
}

// WT[l,r][n][k] = bf16(W_src[l,r,k,n])  (transposed so B-fragment K is contiguous)
__global__ void pack_wt_kernel(const float* __restrict__ W, bf16_t* __restrict__ WT) {
    int idx = blockIdx.x * blockDim.x + threadIdx.x;
    if (idx >= 2 * 8 * 128 * 128) return;
    int lr = idx >> 14;
    int rem = idx & 16383;
    int n = rem >> 7, k = rem & 127;
    WT[idx] = (bf16_t)W[(lr << 14) + (k << 7) + n];
}

// V[l,r][k][h] = sum_j W[l,r,k,h*32+j] * a[l,r,h,j]
__global__ void pack_v_kernel(const float* __restrict__ W, const float* __restrict__ a,
                              float* __restrict__ V) {
    int idx = blockIdx.x * blockDim.x + threadIdx.x;
    if (idx >= 2 * 8 * 128 * 4) return;
    int lr = idx >> 9;
    int rem = idx & 511;
    int k = rem >> 2, h = rem & 3;
    float acc = 0.f;
    for (int j = 0; j < 32; j++)
        acc += W[(lr << 14) + (k << 7) + h * 32 + j] * a[lr * 128 + h * 32 + j];
    V[idx] = acc;
}

__global__ void init_x_kernel(const float* __restrict__ xct, const float* __restrict__ xpet,
                              const float* __restrict__ xclin, const float* __restrict__ xgen,
                              const float* __restrict__ early,
                              float* __restrict__ XC, bf16_t* __restrict__ XBF) {
    size_t idx = (size_t)blockIdx.x * blockDim.x + threadIdx.x;
    size_t total = (size_t)Bsz * NTOT * Dd;
    if (idx >= total) return;
    int k = (int)(idx % Dd);
    size_t bn = idx / Dd;
    int n = (int)(bn % NTOT);
    int b = (int)(bn / NTOT);
    float v;
    if      (n < 1024) v = xct [((size_t)b * 1024 + n)          * Dd + k];
    else if (n < 2048) v = xpet[((size_t)b * 1024 + (n - 1024)) * Dd + k];
    else if (n < 2304) v = xclin[((size_t)b * 256 + (n - 2048)) * Dd + k];
    else if (n < 4352) v = xgen[((size_t)b * 2048 + (n - 2304)) * Dd + k];
    else               v = early[(size_t)b * Dd + k];
    XC[idx] = v;
    XBF[idx] = (bf16_t)v;
}

// ---------- WMMA projection: HS[b,row,:] = XBF[b,soff+row,:] @ Wt ----------
// Weight tile (128x128 bf16 = 32 KB) is staged into LDS once per block with
// gfx1250 async global->LDS copies (ASYNCcnt), then B-fragments come from LDS.
__global__ __launch_bounds__(128)
void proj_wmma_kernel(const bf16_t* __restrict__ XBF, int soff, int sN,
                      const bf16_t* __restrict__ Wt, float* __restrict__ HS) {
    __shared__ __align__(32) bf16_t sW[128 * 128];
    int tid  = threadIdx.x;
    int wave = tid >> 5;
    int lane = tid & 31;

    // ---- async stage: 128 threads x 16 iters x 16 B = 32 KB ----
    {
        unsigned lds0 = (unsigned)(uintptr_t)(&sW[0]);
        unsigned long long g0 = (unsigned long long)(uintptr_t)Wt;
#pragma unroll
        for (int i = 0; i < 16; i++) {
            unsigned loff = (unsigned)(tid + i * 128) * 16u;
            unsigned lds_addr = lds0 + loff;
            unsigned long long gaddr = g0 + loff;
            asm volatile("global_load_async_to_lds_b128 %0, %1, off"
                         :: "v"(lds_addr), "v"(gaddr) : "memory");
        }
        asm volatile("s_wait_asynccnt 0x0" ::: "memory");
    }
    __syncthreads();

    int rowBlocks = (sN + 63) >> 6;
    int b  = blockIdx.x / rowBlocks;
    int rb = blockIdx.x - b * rowBlocks;
    int row0 = rb * 64 + wave * 16;
    int m  = lane & 15;     // A-matrix row within tile / D-matrix column
    int hi = lane >> 4;     // lane half selects K sub-block

    int row = row0 + m;
    if (row >= sN) row = 0;  // clamp loads; stores are guarded
    const bf16_t* xrow = XBF + ((size_t)b * NTOT + soff + row) * Dd;
    __builtin_prefetch(xrow, 0, 3);

    // A fragments: 16x32 bf16. lanes 0-15: K = 32kc+[0..16); lanes 16-31: +16.
    v16bf A[4];
#pragma unroll
    for (int kc = 0; kc < 4; kc++) {
        int kb = kc * 32 + hi * 16;
        v8bf lo = *(const v8bf*)(xrow + kb);
        v8bf up = *(const v8bf*)(xrow + kb + 8);
#pragma unroll
        for (int i = 0; i < 8; i++) { A[kc][i] = lo[i]; A[kc][8 + i] = up[i]; }
    }

    bool full = (row0 + 16) <= sN;

#pragma unroll
    for (int nt = 0; nt < 8; nt++) {
        v8f c = {0.f, 0.f, 0.f, 0.f, 0.f, 0.f, 0.f, 0.f};
#pragma unroll
        for (int kc = 0; kc < 4; kc++) {
            // B fragment 32x16 from LDS: lane -> N = m, K contiguous (sW is [n][k])
            v16bf Bf = *(const v16bf*)(&sW[(size_t)(nt * 16 + m) * Dd + kc * 32 + hi * 16]);
            c = __builtin_amdgcn_wmma_f32_16x16x32_bf16(false, A[kc], false, Bf,
                                                        (short)0, c, false, false);
        }
        if (full) {
            // D layout: VGPR v -> row (row0 + v + 8*hi), column nt*16 + m
            float* basep = HS + ((size_t)b * sN + row0 + hi * 8) * Dd + nt * 16 + m;
#pragma unroll
            for (int v = 0; v < 8; v++) basep[(size_t)v * Dd] = c[v];
        } else {
#pragma unroll
            for (int v = 0; v < 8; v++) {
                int gr = row0 + v + hi * 8;
                if (gr < sN)
                    HS[((size_t)b * sN + gr) * Dd + nt * 16 + m] = c[v];
            }
        }
    }
}

// AL[b,n,h] = XC[b,off+n,:] . V[:,h]
__global__ void att_logits_kernel(const float* __restrict__ XC, int off, int Nn,
                                  const float* __restrict__ V, float* __restrict__ AL) {
    int idx = blockIdx.x * blockDim.x + threadIdx.x;
    if (idx >= Bsz * Nn * 4) return;
    int h = idx & 3;
    int bn = idx >> 2;
    int n = bn % Nn, b = bn / Nn;
    const float* xr = XC + ((size_t)b * NTOT + off + n) * Dd;
    float acc = 0.f;
    for (int k = 0; k < Dd; k++) acc += xr[k] * V[k * 4 + h];
    AL[idx] = acc;
}

// ---------- edge softmax (3 passes) ----------
__global__ void edge_max_kernel(const int* __restrict__ E, const float* __restrict__ ALS,
                                const float* __restrict__ ALD, unsigned* __restrict__ MX,
                                int sN, int dN, int Ecnt) {
    size_t idx = (size_t)blockIdx.x * blockDim.x + threadIdx.x;
    if (idx >= (size_t)Bsz * Ecnt * 4) return;
    int h = (int)(idx & 3);
    size_t be = idx >> 2;
    int e = (int)(be % Ecnt), b = (int)(be / Ecnt);
    int src = E[((size_t)b * 2) * Ecnt + e];
    int dst = E[((size_t)b * 2 + 1) * Ecnt + e];
    float lg = leaky(ALS[((size_t)b * sN + src) * 4 + h] + ALD[((size_t)b * dN + dst) * 4 + h]);
    atomicMax(&MX[((size_t)b * dN + dst) * 4 + h], f2ord(lg));
}

__global__ void edge_den_kernel(const int* __restrict__ E, const float* __restrict__ ALS,
                                const float* __restrict__ ALD, const unsigned* __restrict__ MX,
                                float* __restrict__ DEN, int sN, int dN, int Ecnt) {
    size_t idx = (size_t)blockIdx.x * blockDim.x + threadIdx.x;
    if (idx >= (size_t)Bsz * Ecnt * 4) return;
    int h = (int)(idx & 3);
    size_t be = idx >> 2;
    int e = (int)(be % Ecnt), b = (int)(be / Ecnt);
    int src = E[((size_t)b * 2) * Ecnt + e];
    int dst = E[((size_t)b * 2 + 1) * Ecnt + e];
    float lg = leaky(ALS[((size_t)b * sN + src) * 4 + h] + ALD[((size_t)b * dN + dst) * 4 + h]);
    size_t di = ((size_t)b * dN + dst) * 4 + h;
    atomicAdd(&DEN[di], expf(lg - ord2f(MX[di])));
}

// one wave per (b,edge); lane handles 4 channels (same head)
__global__ void edge_msg_kernel(const int* __restrict__ E, const float* __restrict__ ALS,
                                const float* __restrict__ ALD, const unsigned* __restrict__ MX,
                                const float* __restrict__ DEN, const float* __restrict__ HS,
                                float* __restrict__ AGG, int sN, int dN, int Ecnt, int doff) {
    size_t gw = ((size_t)blockIdx.x * blockDim.x + threadIdx.x) >> 5;
    int lane = threadIdx.x & 31;
    if (gw >= (size_t)Bsz * Ecnt) return;
    int e = (int)(gw % Ecnt), b = (int)(gw / Ecnt);
    int src = E[((size_t)b * 2) * Ecnt + e];
    int dst = E[((size_t)b * 2 + 1) * Ecnt + e];
    int h = lane >> 3;  // channels lane*4..lane*4+3 all belong to head lane/8
    float lg = leaky(ALS[((size_t)b * sN + src) * 4 + h] + ALD[((size_t)b * dN + dst) * 4 + h]);
    size_t di = ((size_t)b * dN + dst) * 4 + h;
    float w = expf(lg - ord2f(MX[di])) / (DEN[di] + 1e-16f);
    const float* hsr = HS + ((size_t)b * sN + src) * Dd + lane * 4;
    float* ag = AGG + ((size_t)b * NTOT + doff + dst) * Dd + lane * 4;
#pragma unroll
    for (int j = 0; j < 4; j++) atomicAdd(&ag[j], w * hsr[j]);
}

__global__ void elu_kernel(const float* __restrict__ AGG, float* __restrict__ XC,
                           bf16_t* __restrict__ XBF) {
    size_t idx = (size_t)blockIdx.x * blockDim.x + threadIdx.x;
    if (idx >= (size_t)Bsz * NTOT * Dd) return;
    float v = AGG[idx];
    float y = v > 0.f ? v : (expf(v) - 1.0f);
    XC[idx] = y;
    XBF[idx] = (bf16_t)y;
}

// ---------- attention pooling: block per (b, type) ----------
__global__ __launch_bounds__(128)
void pool_kernel(const float* __restrict__ XC, const float* __restrict__ Wp,
                 const float* __restrict__ bp, float* __restrict__ POOL) {
    __shared__ float s_mean[128], s_q[128], s_red[128], s_sc[2048];
    int t = threadIdx.x;
    int type = blockIdx.x % 3, b = blockIdx.x / 3;
    const int toff[3] = {0, 1024, 2304};
    const int tn[3]   = {1024, 1024, 2048};
    int off = toff[type], Nn = tn[type];
    const float* feat = XC + ((size_t)b * NTOT + off) * Dd;

    float acc = 0.f;
    for (int n = 0; n < Nn; n++) acc += feat[(size_t)n * Dd + t];
    s_mean[t] = acc / (float)Nn;
    __syncthreads();
    float q = bp[type * Dd + t];
    for (int d = 0; d < Dd; d++) q += s_mean[d] * Wp[((size_t)type * Dd + d) * Dd + t];
    s_q[t] = q;
    __syncthreads();
    float lmax = -1e30f;
    for (int n = t; n < Nn; n += 128) {
        float sc = 0.f;
        for (int d = 0; d < Dd; d++) sc += feat[(size_t)n * Dd + d] * s_q[d];
        s_sc[n] = sc;
        lmax = fmaxf(lmax, sc);
    }
    s_red[t] = lmax; __syncthreads();
    for (int o = 64; o > 0; o >>= 1) { if (t < o) s_red[t] = fmaxf(s_red[t], s_red[t + o]); __syncthreads(); }
    float smax = s_red[0]; __syncthreads();
    float lsum = 0.f;
    for (int n = t; n < Nn; n += 128) { float pz = expf(s_sc[n] - smax); s_sc[n] = pz; lsum += pz; }
    s_red[t] = lsum; __syncthreads();
    for (int o = 64; o > 0; o >>= 1) { if (t < o) s_red[t] += s_red[t + o]; __syncthreads(); }
    float den = s_red[0];
    __syncthreads();
    float o = 0.f;
    for (int n = 0; n < Nn; n++) o += s_sc[n] * feat[(size_t)n * Dd + t];
    POOL[((size_t)type * Bsz + b) * Dd + t] = o / den;
}

// ---------- semantic fusion: block per b ----------
__global__ __launch_bounds__(128)
void semantic_kernel(const float* __restrict__ POOL, const float* __restrict__ Wsem,
                     const float* __restrict__ bsem, const float* __restrict__ qsem,
                     float* __restrict__ out) {
    __shared__ float z[3][256];
    __shared__ float s_red[128];
    __shared__ float s_s[3];
    __shared__ float s_beta[3];
    int t = threadIdx.x, b = blockIdx.x;
    float ct = POOL[(0 * Bsz + b) * 128 + t];
    float pe = POOL[(1 * Bsz + b) * 128 + t];
    float ge = POOL[(2 * Bsz + b) * 128 + t];
    z[0][t] = ct;               z[0][128 + t] = ge;
    z[1][t] = pe;               z[1][128 + t] = ge;
    z[2][t] = 0.5f * (ct + pe); z[2][128 + t] = ge;
    __syncthreads();
    for (int p = 0; p < 3; p++) {
        float acc = bsem[t];
        for (int f = 0; f < 256; f++) acc += z[p][f] * Wsem[f * 128 + t];  // z[f>=256]==0
        float v = qsem[t] * tanhf(acc);
        s_red[t] = v; __syncthreads();
        for (int o = 64; o > 0; o >>= 1) { if (t < o) s_red[t] += s_red[t + o]; __syncthreads(); }
        if (t == 0) s_s[p] = s_red[0];
        __syncthreads();
    }
    if (t == 0) {
        float m = fmaxf(s_s[0], fmaxf(s_s[1], s_s[2]));
        float e0 = expf(s_s[0] - m), e1 = expf(s_s[1] - m), e2 = expf(s_s[2] - m);
        float d = e0 + e1 + e2;
        s_beta[0] = e0 / d; s_beta[1] = e1 / d; s_beta[2] = e2 / d;
    }
    __syncthreads();
    for (int f = t; f < 512; f += 128) {
        float v = 0.f;
        if (f < 256) v = s_beta[0] * z[0][f] + s_beta[1] * z[1][f] + s_beta[2] * z[2][f];
        out[(size_t)b * 512 + f] = v;
    }
    if (t < 3) out[(size_t)Bsz * 512 + b * 3 + t] = s_beta[t];
}

// ---------- host ----------
extern "C" void kernel_launch(void* const* d_in, const int* in_sizes, int n_in,
                              void* d_out, int out_size, void* d_ws, size_t ws_size,
                              hipStream_t stream) {
    (void)in_sizes; (void)n_in; (void)out_size; (void)ws_size;
    const float* x_ct  = (const float*)d_in[0];
    const float* x_pet = (const float*)d_in[1];
    const float* x_clin= (const float*)d_in[2];
    const float* x_gen = (const float*)d_in[3];
    const float* early = (const float*)d_in[4];
    const int* eptr[8];
    for (int i = 0; i < 8; i++) eptr[i] = (const int*)d_in[5 + i];
    const float* W_src = (const float*)d_in[13];
    const float* W_dst = (const float*)d_in[14];
    const float* a_src = (const float*)d_in[15];
    const float* a_dst = (const float*)d_in[16];
    const float* Wp    = (const float*)d_in[17];
    const float* bp    = (const float*)d_in[18];
    const float* W_sem = (const float*)d_in[19];
    const float* b_sem = (const float*)d_in[20];
    const float* q_sem = (const float*)d_in[21];

    char* pw = (char*)d_ws;
    auto carve = [&](size_t bytes) -> void* {
        void* r = (void*)pw;
        pw += (bytes + 255) & ~(size_t)255;
        return r;
    };
    const size_t SZX = (size_t)Bsz * NTOT * Dd;
    bf16_t*  WT  = (bf16_t*)carve((size_t)2 * 8 * 16384 * sizeof(bf16_t));
    float*   VS  = (float*) carve((size_t)2 * 8 * 512 * sizeof(float));
    float*   VD  = (float*) carve((size_t)2 * 8 * 512 * sizeof(float));
    float*   XC  = (float*) carve(SZX * sizeof(float));
    bf16_t*  XBF = (bf16_t*)carve(SZX * sizeof(bf16_t));
    float*   AGG = (float*) carve(SZX * sizeof(float));
    float*   HS  = (float*) carve((size_t)Bsz * 2048 * Dd * sizeof(float));
    float*   ALS = (float*) carve((size_t)Bsz * 2048 * 4 * sizeof(float));
    float*   ALD = (float*) carve((size_t)Bsz * 2048 * 4 * sizeof(float));
    unsigned* MX = (unsigned*)carve((size_t)Bsz * 2048 * 4 * sizeof(unsigned));
    float*   DEN = (float*) carve((size_t)Bsz * 2048 * 4 * sizeof(float));
    float*  POOL = (float*) carve((size_t)3 * Bsz * Dd * sizeof(float));

    pack_wt_kernel<<<(2 * 8 * 16384 + 255) / 256, 256, 0, stream>>>(W_src, WT);
    pack_v_kernel <<<(2 * 8 * 512 + 255) / 256, 256, 0, stream>>>(W_src, a_src, VS);
    pack_v_kernel <<<(2 * 8 * 512 + 255) / 256, 256, 0, stream>>>(W_dst, a_dst, VD);
    init_x_kernel <<<(unsigned)((SZX + 255) / 256), 256, 0, stream>>>(
        x_ct, x_pet, x_clin, x_gen, early, XC, XBF);

    // relation tables: (src_off, src_N, dst_off, dst_N, E)
    const int soffs[8] = {0, 1024, 2048, 2304, 4352, 0, 1024, 2304};
    const int sNs[8]   = {1024, 1024, 256, 2048, 1, 1024, 1024, 2048};
    const int doffs[8] = {0, 1024, 2048, 2304, 2048, 2304, 2304, 4352};
    const int dNs[8]   = {1024, 1024, 256, 2048, 256, 2048, 2048, 1};
    const int Es[8]    = {8192, 8192, 2048, 16384, 256, 8192, 8192, 2048};

    for (int l = 0; l < 2; l++) {
        zero_f32_kernel<<<(unsigned)((SZX + 255) / 256), 256, 0, stream>>>(AGG, SZX);
        for (int r = 0; r < 8; r++) {
            int sN = sNs[r], dN = dNs[r], E = Es[r];
            int rowBlocks = (sN + 63) / 64;
            proj_wmma_kernel<<<(unsigned)(Bsz * rowBlocks), 128, 0, stream>>>(
                XBF, soffs[r], sN, WT + (size_t)(l * 8 + r) * 16384, HS);
            att_logits_kernel<<<(unsigned)(((size_t)Bsz * sN * 4 + 255) / 256), 256, 0, stream>>>(
                XC, soffs[r], sN, VS + (size_t)(l * 8 + r) * 512, ALS);
            att_logits_kernel<<<(unsigned)(((size_t)Bsz * dN * 4 + 255) / 256), 256, 0, stream>>>(
                XC, doffs[r], dN, VD + (size_t)(l * 8 + r) * 512, ALD);
            size_t mdcnt = (size_t)Bsz * dN * 4;
            zero_u32_kernel<<<(unsigned)((mdcnt + 255) / 256), 256, 0, stream>>>(MX, mdcnt);
            zero_f32_kernel<<<(unsigned)((mdcnt + 255) / 256), 256, 0, stream>>>(DEN, mdcnt);
            size_t ethreads = (size_t)Bsz * E * 4;
            edge_max_kernel<<<(unsigned)((ethreads + 255) / 256), 256, 0, stream>>>(
                eptr[r], ALS, ALD, MX, sN, dN, E);
            edge_den_kernel<<<(unsigned)((ethreads + 255) / 256), 256, 0, stream>>>(
                eptr[r], ALS, ALD, MX, DEN, sN, dN, E);
            size_t mthreads = (size_t)Bsz * E * 32;  // one wave per edge
            edge_msg_kernel<<<(unsigned)((mthreads + 255) / 256), 256, 0, stream>>>(
                eptr[r], ALS, ALD, MX, DEN, HS, AGG, sN, dN, E, doffs[r]);
        }
        elu_kernel<<<(unsigned)((SZX + 255) / 256), 256, 0, stream>>>(AGG, XC, XBF);
    }

    pool_kernel<<<Bsz * 3, 128, 0, stream>>>(XC, Wp, bp, POOL);
    semantic_kernel<<<Bsz, 128, 0, stream>>>(POOL, W_sem, b_sem, q_sem, (float*)d_out);
}
